// FusionAwareInterpCVPR_37795712204987
// MI455X (gfx1250) — matrix operand order repaired and checked by reading, not compiled, not run on hardware
//
#include <hip/hip_runtime.h>
#include <hip/hip_bf16.h>

// ---------------------------------------------------------------------------
// FusionAwareInterp for MI455X (gfx1250, wave32, WMMA + TDM)
// B=4, C=128, H=W=64, HW=N=4096
// ---------------------------------------------------------------------------

#define Bn   4
#define Cn   128
#define Hn   64
#define Wn   64
#define HWn  4096
#define Nn   4096
#define K1P  160   // 131 padded to multiple of 32 (WMMA K-step)
#define K2P  128

typedef __attribute__((ext_vector_type(16))) _Float16 v16h;
typedef __attribute__((ext_vector_type(8)))  float    v8f;
typedef __attribute__((ext_vector_type(4)))  unsigned uint32x4;
typedef __attribute__((ext_vector_type(8)))  int      int32x8;
typedef __attribute__((ext_vector_type(4)))  int      int32x4;

union H16 { uint4 u[2]; v16h v; };

// ---------------------------------------------------------------------------
// Kernel 1: brute-force nearest neighbor.  uv (per batch, 2x4096) staged in
// LDS (32 KB of the 320 KB/WGP pool); one thread per grid point.
// ---------------------------------------------------------------------------
__global__ __launch_bounds__(256) void nn_kernel(const float* __restrict__ uv,
                                                 int* __restrict__ nn_idx)
{
    __shared__ float sux[Nn];
    __shared__ float suy[Nn];
    const int b   = blockIdx.y;
    const int tid = threadIdx.x;
    const float* uvb = uv + (size_t)b * 2 * Nn;
    for (int i = tid; i < Nn; i += 256) {
        sux[i] = uvb[i];
        suy[i] = uvb[Nn + i];
    }
    __syncthreads();

    const int q = blockIdx.x * 256 + tid;
    const float gx = (float)(q & 63);
    const float gy = (float)(q >> 6);
    float best = 3.0e38f;
    int   bi   = 0;
    #pragma unroll 4
    for (int n = 0; n < Nn; ++n) {
        const float dx = gx - sux[n];
        const float dy = gy - suy[n];
        const float d  = dx * dx + dy * dy;
        if (d < best) { best = d; bi = n; }   // strict < keeps first index (argmin tie rule)
    }
    nn_idx[b * HWn + q] = bi;
}

// ---------------------------------------------------------------------------
// Kernel 2: fused gather + bilinear + correlation -> 160-channel padded feat,
// stored directly in f16 (halves intermediate traffic; consumed as f16 by the
// WMMA anyway).  feat channels: 0,1 = uv_nn - grid ; 2 = corr ; 3..130 =
// feat3d[:, nn] ; 131..159 = 0 (K padding).
// ---------------------------------------------------------------------------
__global__ __launch_bounds__(256) void feat_kernel(const float* __restrict__ uv,
                                                   const float* __restrict__ feat2d,
                                                   const float* __restrict__ feat3d,
                                                   const int*   __restrict__ nn_idx,
                                                   _Float16* __restrict__ feat)
{
    const int b = blockIdx.y;
    const int q = blockIdx.x * 256 + threadIdx.x;
    const int n = nn_idx[b * HWn + q];

    const float* uvb = uv + (size_t)b * 2 * Nn;
    const float ux = uvb[n];
    const float uy = uvb[Nn + n];

    const float x0 = floorf(ux), y0 = floorf(uy);
    const float wx = ux - x0,    wy = uy - y0;
    int x0i = min(max((int)x0, 0), Wn - 1);
    int x1i = min(x0i + 1, Wn - 1);
    int y0i = min(max((int)y0, 0), Hn - 1);
    int y1i = min(y0i + 1, Hn - 1);

    const float w00 = (1.f - wx) * (1.f - wy);
    const float w01 = wx * (1.f - wy);
    const float w10 = (1.f - wx) * wy;
    const float w11 = wx * wy;

    const int i00 = y0i * Wn + x0i, i01 = y0i * Wn + x1i;
    const int i10 = y1i * Wn + x0i, i11 = y1i * Wn + x1i;

    const float* f2b = feat2d + (size_t)b * Cn * HWn;
    const float* f3b = feat3d + (size_t)b * Cn * Nn;
    _Float16*    fb  = feat   + (size_t)b * K1P * HWn;

    float corr = 0.f;
    #pragma unroll 4
    for (int c = 0; c < Cn; ++c) {
        const float* f2c = f2b + (size_t)c * HWn;
        const float s = f2c[i00] * w00 + f2c[i01] * w01 +
                        f2c[i10] * w10 + f2c[i11] * w11;
        corr += s * f2c[q];
        fb[(size_t)(3 + c) * HWn + q] = (_Float16)f3b[(size_t)c * Nn + n];
    }
    fb[q]           = (_Float16)(ux - (float)(q & 63));
    fb[HWn + q]     = (_Float16)(uy - (float)(q >> 6));
    fb[2 * HWn + q] = (_Float16)(corr * (1.0f / (float)Cn));
    #pragma unroll
    for (int c = 131; c < K1P; ++c) fb[(size_t)c * HWn + q] = (_Float16)0.f;
}

// ---------------------------------------------------------------------------
// Kernel 3: convert weights to f16 (w1 K-padded 131 -> 160).
// ---------------------------------------------------------------------------
__global__ __launch_bounds__(256) void prep_weights(const float* __restrict__ w1,
                                                    const float* __restrict__ w2,
                                                    const float* __restrict__ w3,
                                                    _Float16* __restrict__ w1h,
                                                    _Float16* __restrict__ w2h,
                                                    _Float16* __restrict__ w3h)
{
    const int idx = blockIdx.x * 256 + threadIdx.x;
    if (idx < Cn * K1P) {
        const int o = idx / K1P, c = idx % K1P;
        w1h[idx] = (_Float16)(c < 131 ? w1[o * 131 + c] : 0.f);
    }
    const int j = idx - Cn * K1P;
    if (j >= 0 && j < Cn * Cn) {
        w2h[j] = (_Float16)w2[j];
        w3h[j] = (_Float16)w3[j];
    }
}

// ---------------------------------------------------------------------------
// Kernel 4: GEMM + bias + leaky-ReLU via v_wmma_f32_16x16x32_f16.
// Y[b,o,p] = lrelu( sum_k Wh[o,k] * X[b,k,p] + bias[o] )
// Block: 256 thr = 8 waves; wave w owns the 16-row M-tile [16w,16w+16) of one
// 16-column N-tile.  The full f16 weight matrix (Cn x KP, contiguous, <=40KB)
// is staged into LDS by the Tensor Data Mover (wave 0 issues a 1-D-tile D#,
// waits on TENSORcnt, then the workgroup barrier publishes it).  The X tile
// is staged transposed to f16 LDS so each lane's B fragment (K = grp*8+0..7, 16+grp*8+0..7)
// is two runs of 8 contiguous halves (two ds_load_b128).
// ---------------------------------------------------------------------------
template<int KP, typename OutT>
__global__ __launch_bounds__(256) void gemm16_lrelu(const _Float16* __restrict__ Wh,
                                                    const _Float16* __restrict__ X,
                                                    const float* __restrict__ bias,
                                                    OutT* __restrict__ Y)
{
    __shared__ __attribute__((aligned(16))) _Float16 ldsW[Cn * KP];   // weights
    __shared__ __attribute__((aligned(16))) _Float16 ldsXT[16 * KP];  // [n][k]

    const int b   = blockIdx.y;
    const int p0  = blockIdx.x * 16;
    const int tid = threadIdx.x;

#if __has_builtin(__builtin_amdgcn_tensor_load_to_lds)
    // Wave 0: TDM copy of the whole weight matrix (contiguous Cn*KP halves).
    if (tid < 32) {
        const unsigned long long ga = (unsigned long long)(const void*)Wh;
        const unsigned lds_off = (unsigned)(unsigned long long)(const void*)&ldsW[0];
        const unsigned NT = (unsigned)(Cn * KP);            // halves, < 65536
        // D# group 0: count=1 | lds_addr | global_addr[56:0] | type=2
        uint32x4 g0 = { 1u,
                        lds_off,
                        (unsigned)(ga & 0xFFFFFFFFull),
                        (unsigned)((ga >> 32) & 0x01FFFFFFull) | 0x80000000u };
        // D# group 1: data_size=1(2B); tensor_dim0=NT, tensor_dim1=1;
        //             tile_dim0=NT, tile_dim1=1; tensor_dim0_stride=NT.
        int32x8 g1 = { (int)0x00010000u,                          // data_size=2B
                       (int)((NT & 0xFFFFu) << 16),               // tensor_dim0[15:0]
                       (int)(((NT >> 16) & 0xFFFFu) | (1u << 16)),// tensor_dim0[31:16] | tensor_dim1=1
                       (int)((NT & 0xFFFFu) << 16),               // tile_dim0 = NT
                       1,                                         // tile_dim1 = 1
                       (int)NT,                                   // tensor_dim0_stride
                       0, 0 };
        int32x4 gz = { 0, 0, 0, 0 };
#if __clang_major__ >= 23
        int32x8 gz8 = { 0, 0, 0, 0, 0, 0, 0, 0 };
        __builtin_amdgcn_tensor_load_to_lds(g0, g1, gz, gz, gz8, 0);
#else
        __builtin_amdgcn_tensor_load_to_lds(g0, g1, gz, gz, 0);
#endif
        __builtin_amdgcn_s_wait_tensorcnt(0);
    }
#else
    for (int idx = tid; idx < Cn * KP; idx += 256) ldsW[idx] = Wh[idx];
#endif

    // Stage + transpose the f16 X tile (KP x 16) into LDS as [n][k].
    const _Float16* Xb = X + (size_t)b * KP * HWn;
    for (int idx = tid; idx < KP * 16; idx += 256) {
        const int k = idx >> 4;
        const int n = idx & 15;
        ldsXT[n * KP + k] = Xb[(size_t)k * HWn + p0 + n];
    }
    __syncthreads();

    const int wave = tid >> 5;        // wave32
    const int lane = tid & 31;
    const int grp  = lane >> 4;       // 0: lanes 0-15, 1: lanes 16-31
    const int nm   = lane & 15;

    const _Float16* wrow = ldsW  + (size_t)(wave * 16 + nm) * KP; // A row M
    const _Float16* xcol = ldsXT + (size_t)nm * KP;               // B col N

    v8f acc = {};
    #pragma unroll
    for (int k0 = 0; k0 < KP; k0 += 32) {
        // A fragment: halves 0..7 -> K=k0+grp*8+0..7 ; 8..15 -> K=k0+16+grp*8+0..7
        H16 a;
        a.u[0] = *(const uint4*)(wrow + k0 + grp * 8);
        a.u[1] = *(const uint4*)(wrow + k0 + 16 + grp * 8);
        // B fragment: halves 0..7 -> K=k0+grp*8+0..7 ; 8..15 -> K=k0+16+grp*8+0..7
        H16 bm;
        bm.u[0] = *(const uint4*)(xcol + k0 + grp * 8);
        bm.u[1] = *(const uint4*)(xcol + k0 + 16 + grp * 8);

        acc = __builtin_amdgcn_wmma_f32_16x16x32_f16(
            /*neg_a=*/false, a.v, /*neg_b=*/false, bm.v,
            /*c_mod=*/(short)0, acc, /*reuse_a=*/false, /*reuse_b=*/false);
    }

    // D layout: acc[j] -> row wave*16 + j + 8*grp, col p0 + nm.
    const int col = p0 + nm;
    OutT* Yb = Y + (size_t)b * Cn * HWn;
    #pragma unroll
    for (int j = 0; j < 8; ++j) {
        const int o = wave * 16 + j + grp * 8;
        float y = acc[j] + bias[o];
        y = (y >= 0.f) ? y : 0.1f * y;
        Yb[(size_t)o * HWn + col] = (OutT)y;
    }
}

// ---------------------------------------------------------------------------
// Launch
// ---------------------------------------------------------------------------
extern "C" void kernel_launch(void* const* d_in, const int* in_sizes, int n_in,
                              void* d_out, int out_size, void* d_ws, size_t ws_size,
                              hipStream_t stream)
{
    const float* uv     = (const float*)d_in[0];
    const float* feat2d = (const float*)d_in[1];
    const float* feat3d = (const float*)d_in[2];
    const float* w1     = (const float*)d_in[3];
    const float* b1     = (const float*)d_in[4];
    const float* w2     = (const float*)d_in[5];
    const float* b2     = (const float*)d_in[6];
    const float* w3     = (const float*)d_in[7];
    const float* b3     = (const float*)d_in[8];
    float* out = (float*)d_out;

    char* ws = (char*)d_ws;
    int*      nn   = (int*)ws;                     //    65536 B
    _Float16* w1h  = (_Float16*)(ws + 65536);      //    40960 B (128*160)
    _Float16* w2h  = (_Float16*)(ws + 106496);     //    32768 B
    _Float16* w3h  = (_Float16*)(ws + 139264);     //    32768 B
    _Float16* feat = (_Float16*)(ws + 172032);     //  5242880 B (4*160*4096 f16)
    _Float16* x1   = (_Float16*)(ws + 5414912);    //  4194304 B (4*128*4096 f16)
    _Float16* x2   = feat;  // reuse: feat dead after conv1; region is big enough

    nn_kernel   <<<dim3(HWn / 256, Bn), 256, 0, stream>>>(uv, nn);
    prep_weights<<<(Cn * K1P + Cn * Cn + 255) / 256, 256, 0, stream>>>(w1, w2, w3, w1h, w2h, w3h);
    feat_kernel <<<dim3(HWn / 256, Bn), 256, 0, stream>>>(uv, feat2d, feat3d, nn, feat);

    gemm16_lrelu<K1P, _Float16><<<dim3(HWn / 16, Bn), 256, 0, stream>>>(w1h, feat, b1, x1);
    gemm16_lrelu<K2P, _Float16><<<dim3(HWn / 16, Bn), 256, 0, stream>>>(w2h, x1,  b2, x2);
    gemm16_lrelu<K2P, float>   <<<dim3(HWn / 16, Bn), 256, 0, stream>>>(w3h, x2,  b3, out);
}
